// MambaBlock_85950885528371
// MI455X (gfx1250) — matrix operand order repaired and checked by reading, not compiled, not run on hardware
//
#include <hip/hip_runtime.h>
#include <hip/hip_bf16.h>

typedef _Float16 v8h  __attribute__((ext_vector_type(8)));
typedef _Float16 v16h __attribute__((ext_vector_type(16)));
typedef float    v8f  __attribute__((ext_vector_type(8)));

#define B_  4
#define L_  2048
#define D_  1024
#define DI_ 1024
#define N_  8
#define K_  4
#define R_  64
#define M_  (B_*L_)     // 8192 rows
#define CH_ 16          // scan chunks
#define LC_ (L_/CH_)    // 128 steps per chunk

// ---------------------------------------------------------------- utilities
__global__ void cvt_f32_f16(const float* __restrict__ src, _Float16* __restrict__ dst, int n) {
    int i = blockIdx.x * blockDim.x + threadIdx.x;
    if (i < n) dst[i] = (_Float16)src[i];
}

// RMSNorm over D=1024, one block per row, writes f16 for the WMMA GEMM.
__global__ void rmsnorm_f16(const float* __restrict__ x, const float* __restrict__ w,
                            _Float16* __restrict__ out) {
    __shared__ float red[256];
    const int row = blockIdx.x;
    const float* xr = x + (size_t)row * D_;
    float s = 0.f;
    for (int i = threadIdx.x; i < D_; i += 256) { float v = xr[i]; s += v * v; }
    red[threadIdx.x] = s;
    __syncthreads();
    for (int off = 128; off > 0; off >>= 1) {
        if (threadIdx.x < off) red[threadIdx.x] += red[threadIdx.x + off];
        __syncthreads();
    }
    const float scale = rsqrtf(red[0] * (1.f / (float)D_) + 1e-5f);
    _Float16* orow = out + (size_t)row * D_;
    for (int i = threadIdx.x; i < D_; i += 256)
        orow[i] = (_Float16)(xr[i] * scale * w[i]);
}

// ------------------------------------------------- WMMA GEMM: C = A * W^T
// A: [M, K] f16 row-major. W: [N, K] f16 row-major (B-matrix[k][n] = W[n][k]).
// Block = 256 threads (8 waves). All 8 waves share one 16-wide N-tile whose
// weight fragments are staged once in LDS in exact per-lane register layout.
// Inner loop is register double-buffered: fragment ks+1 is loaded before the
// WMMA on fragment ks so global/ds loads overlap the matrix op.
// EPILOG: 0 = plain f32 store, 1 = softplus(v + bias[col]).
template <int EPILOG>
__global__ void gemm_wmma_f16(const _Float16* __restrict__ A, const _Float16* __restrict__ W,
                              float* __restrict__ C, const float* __restrict__ bias,
                              int Kdim, int Ncols) {
    extern __shared__ char smem[];          // ksteps * 1024 bytes of B fragments
    const int lane  = threadIdx.x & 31;
    const int wave  = threadIdx.x >> 5;
    const int nloc  = lane & 15;
    const int khalf = lane >> 4;
    const int ksteps = Kdim >> 5;
    const int col = blockIdx.x * 16 + nloc;

    // Stage weight fragments: lane (n=nloc, khalf) needs W[col][ks*32+khalf*16 .. +15],
    // a contiguous 32B chunk -> smem[ks*1024 + lane*32].
    {
        const _Float16* wrow = W + (size_t)col * Kdim + khalf * 16;
        for (int ks = wave; ks < ksteps; ks += 8) {
            const v8h* src = (const v8h*)(wrow + ks * 32);
            v8h b0 = src[0];
            v8h b1 = src[1];
            v8h* dst = (v8h*)(smem + (size_t)ks * 1024 + lane * 32);
            dst[0] = b0;
            dst[1] = b1;
        }
    }
    __syncthreads();

    const int tm = blockIdx.y * 8 + wave;                 // 16-row tile index
    const _Float16* abase = A + (size_t)(tm * 16 + nloc) * Kdim + khalf * 16;

    v8f acc = {};
    // prologue: fragment 0 in registers
    v8h a0 = *(const v8h*)(abase);
    v8h a1 = *(const v8h*)(abase + 8);
    const v8h* bp0 = (const v8h*)(smem + lane * 32);
    v8h b0 = bp0[0];
    v8h b1 = bp0[1];

    for (int ks = 0; ks < ksteps; ++ks) {
        const int kn = (ks + 1 < ksteps) ? (ks + 1) : ks;   // clamped next
        v8h na0 = *(const v8h*)(abase + kn * 32);
        v8h na1 = *(const v8h*)(abase + kn * 32 + 8);
        const v8h* nbp = (const v8h*)(smem + (size_t)kn * 1024 + lane * 32);
        v8h nb0 = nbp[0];
        v8h nb1 = nbp[1];
        if (ks + 2 < ksteps)
            __builtin_prefetch(abase + (ks + 2) * 32, 0, 0);   // global_prefetch_b8

        v16h a = __builtin_shufflevector(a0, a1, 0,1,2,3,4,5,6,7,8,9,10,11,12,13,14,15);
        v16h b = __builtin_shufflevector(b0, b1, 0,1,2,3,4,5,6,7,8,9,10,11,12,13,14,15);
        acc = __builtin_amdgcn_wmma_f32_16x16x32_f16(false, a, false, b,
                                                     (short)0, acc, false, false);
        a0 = na0; a1 = na1; b0 = nb0; b1 = nb1;
    }

    // C/D layout: vgpr r -> row m = r + 8*khalf, col n = nloc
    const int rbase = tm * 16 + khalf * 8;
    float bv = 0.f;
    if (EPILOG == 1) bv = bias[col];
    #pragma unroll
    for (int r = 0; r < 8; ++r) {
        float v = acc[r];
        if (EPILOG == 1) {
            v += bv;
            v = (v > 20.f) ? v : log1pf(__expf(v));       // softplus
        }
        C[(size_t)(rbase + r) * Ncols + col] = v;
    }
}

// ----------------------------------------- causal depthwise conv (K=4) + SiLU
__global__ void conv_silu(const float* __restrict__ xz, const float* __restrict__ cw,
                          const float* __restrict__ cb,
                          float* __restrict__ xc, _Float16* __restrict__ xc16) {
    int idx = blockIdx.x * blockDim.x + threadIdx.x;      // over M_*DI_
    if (idx >= M_ * DI_) return;
    const int d = idx & (DI_ - 1);
    const int m = idx >> 10;
    const int l = m & (L_ - 1);
    const float* base = xz + (size_t)m * (2 * DI_) + d;
    float acc = cb[d];
    #pragma unroll
    for (int j = 0; j < K_; ++j) {
        int ll = l + j - (K_ - 1);
        if (ll >= 0) acc += cw[d * K_ + j] * base[(j - (K_ - 1)) * (2 * DI_)];
    }
    float s = acc / (1.f + __expf(-acc));                 // silu
    xc[idx]   = s;
    xc16[idx] = (_Float16)s;
}

// Slice first R=64 columns of dbl [M, 80] to f16 for the dt_proj GEMM.
__global__ void prep_dt16(const float* __restrict__ dbl, _Float16* __restrict__ dt16) {
    int idx = blockIdx.x * blockDim.x + threadIdx.x;      // M_*R_
    if (idx >= M_ * R_) return;
    int m = idx >> 6, r = idx & 63;
    dt16[idx] = (_Float16)dbl[(size_t)m * (R_ + 2 * N_) + r];
}

// --------------------------- chunked selective scan (3-pass decoupled scan)
// Recurrence: h <- exp(dt*A[n])*h + dt*B*x  (diagonal), so a chunk's transition
// is exp(A[n] * sum(dt)) -- one scalar sum(dt) per (b,d,chunk) suffices.

// Pass A: per (b,d,chunk) local scan with h0=0; emit chunk-end state S and sum(dt).
__global__ void scan_partial(const float* __restrict__ dt, const float* __restrict__ xc,
                             const float* __restrict__ dbl, const float* __restrict__ A_log,
                             float* __restrict__ chunkS, float* __restrict__ chunkSum) {
    const int idx = blockIdx.x * blockDim.x + threadIdx.x;   // B_*CH_*DI_
    const int d = idx & (DI_ - 1);
    const int t = idx >> 10;
    const int c = t & (CH_ - 1);
    const int b = t >> 4;
    float A[N_];
    #pragma unroll
    for (int n = 0; n < N_; ++n) A[n] = -__expf(A_log[d * N_ + n]);
    float h[N_] = {};
    float sumdt = 0.f;
    for (int l = c * LC_; l < (c + 1) * LC_; ++l) {
        const size_t m = (size_t)b * L_ + l;
        const float dtv = dt[m * DI_ + d];
        const float xv  = xc[m * DI_ + d];
        const float* bc = dbl + m * (R_ + 2 * N_) + R_;
        sumdt += dtv;
        #pragma unroll
        for (int n = 0; n < N_; ++n)
            h[n] = __expf(dtv * A[n]) * h[n] + dtv * bc[n] * xv;
    }
    const size_t slot = ((size_t)(b * DI_ + d) * CH_ + c);
    #pragma unroll
    for (int n = 0; n < N_; ++n) chunkS[slot * N_ + n] = h[n];
    chunkSum[slot] = sumdt;
}

// Pass B: per (b,d) combine across the 16 chunks -> initial state per chunk.
__global__ void scan_combine(const float* __restrict__ chunkS, const float* __restrict__ chunkSum,
                             const float* __restrict__ A_log, float* __restrict__ hinit) {
    const int idx = blockIdx.x * blockDim.x + threadIdx.x;   // B_*DI_
    const int d = idx & (DI_ - 1);
    float A[N_];
    #pragma unroll
    for (int n = 0; n < N_; ++n) A[n] = -__expf(A_log[d * N_ + n]);
    float h[N_] = {};
    const size_t base = (size_t)idx * CH_;
    for (int c = 0; c < CH_; ++c) {
        #pragma unroll
        for (int n = 0; n < N_; ++n) hinit[(base + c) * N_ + n] = h[n];
        const float sdt = chunkSum[base + c];
        #pragma unroll
        for (int n = 0; n < N_; ++n)
            h[n] = __expf(A[n] * sdt) * h[n] + chunkS[(base + c) * N_ + n];
    }
}

// Pass C: per (b,d,chunk) re-run from true initial state; emit gated f16 output.
__global__ void scan_final(const float* __restrict__ dt, const float* __restrict__ xc,
                           const float* __restrict__ dbl, const float* __restrict__ xz,
                           const float* __restrict__ A_log, const float* __restrict__ Dskip,
                           const float* __restrict__ hinit, _Float16* __restrict__ y16) {
    const int idx = blockIdx.x * blockDim.x + threadIdx.x;   // B_*CH_*DI_
    const int d = idx & (DI_ - 1);
    const int t = idx >> 10;
    const int c = t & (CH_ - 1);
    const int b = t >> 4;
    float A[N_];
    #pragma unroll
    for (int n = 0; n < N_; ++n) A[n] = -__expf(A_log[d * N_ + n]);
    float h[N_];
    const size_t slot = ((size_t)(b * DI_ + d) * CH_ + c);
    #pragma unroll
    for (int n = 0; n < N_; ++n) h[n] = hinit[slot * N_ + n];
    const float dsk = Dskip[d];
    for (int l = c * LC_; l < (c + 1) * LC_; ++l) {
        const size_t m = (size_t)b * L_ + l;
        const float dtv = dt[m * DI_ + d];
        const float xv  = xc[m * DI_ + d];
        const float* bc = dbl + m * (R_ + 2 * N_) + R_;
        float y = 0.f;
        #pragma unroll
        for (int n = 0; n < N_; ++n) {
            float hn = __expf(dtv * A[n]) * h[n] + dtv * bc[n] * xv;
            h[n] = hn;
            y += hn * bc[N_ + n];
        }
        y += dsk * xv;
        const float z = xz[m * (2 * DI_) + DI_ + d];
        y *= z / (1.f + __expf(-z));                          // * silu(z)
        y16[m * DI_ + d] = (_Float16)y;
    }
}

// ---------------------------------------------------------------- launcher
extern "C" void kernel_launch(void* const* d_in, const int* in_sizes, int n_in,
                              void* d_out, int out_size, void* d_ws, size_t ws_size,
                              hipStream_t stream) {
    const float* hidden    = (const float*)d_in[0];
    const float* norm_w    = (const float*)d_in[1];
    const float* in_proj_w = (const float*)d_in[2];   // [2*DI, D]
    const float* conv_w    = (const float*)d_in[3];   // [DI, 1, K]
    const float* conv_b    = (const float*)d_in[4];
    const float* x_proj_w  = (const float*)d_in[5];   // [R+2N, DI]
    const float* dt_proj_w = (const float*)d_in[6];   // [DI, R]
    const float* dt_proj_b = (const float*)d_in[7];
    const float* A_log     = (const float*)d_in[8];   // [DI, N]
    const float* D_skip    = (const float*)d_in[9];
    const float* out_proj_w= (const float*)d_in[10];  // [D, DI]
    float* out = (float*)d_out;

    char* ws = (char*)d_ws;
    size_t o = 0;
    auto take = [&](size_t bytes) { char* p = ws + o; o += (bytes + 255) & ~(size_t)255; return p; };
    _Float16* x16     = (_Float16*)take((size_t)M_ * D_ * 2);
    _Float16* w16_in  = (_Float16*)take((size_t)2 * DI_ * D_ * 2);
    _Float16* w16_xp  = (_Float16*)take((size_t)(R_ + 2 * N_) * DI_ * 2);
    _Float16* w16_dt  = (_Float16*)take((size_t)DI_ * R_ * 2);
    _Float16* w16_out = (_Float16*)take((size_t)D_ * DI_ * 2);
    float*    xz      = (float*)   take((size_t)M_ * 2 * DI_ * 4);
    float*    xc      = (float*)   take((size_t)M_ * DI_ * 4);
    _Float16* xc16    = (_Float16*)take((size_t)M_ * DI_ * 2);
    float*    dbl     = (float*)   take((size_t)M_ * (R_ + 2 * N_) * 4);
    _Float16* dt16    = (_Float16*)take((size_t)M_ * R_ * 2);
    float*    dtbuf   = (float*)   take((size_t)M_ * DI_ * 4);
    _Float16* y16     = (_Float16*)take((size_t)M_ * DI_ * 2);
    float*    chunkS  = (float*)   take((size_t)B_ * DI_ * CH_ * N_ * 4);
    float*    chunkSum= (float*)   take((size_t)B_ * DI_ * CH_ * 4);
    float*    hinit   = (float*)   take((size_t)B_ * DI_ * CH_ * N_ * 4);

    // Weight down-conversion to f16
    {
        int n;
        n = 2 * DI_ * D_;        cvt_f32_f16<<<(n + 255) / 256, 256, 0, stream>>>(in_proj_w,  w16_in,  n);
        n = (R_ + 2 * N_) * DI_; cvt_f32_f16<<<(n + 255) / 256, 256, 0, stream>>>(x_proj_w,   w16_xp,  n);
        n = DI_ * R_;            cvt_f32_f16<<<(n + 255) / 256, 256, 0, stream>>>(dt_proj_w,  w16_dt,  n);
        n = D_ * DI_;            cvt_f32_f16<<<(n + 255) / 256, 256, 0, stream>>>(out_proj_w, w16_out, n);
    }

    // 1) RMSNorm -> f16
    rmsnorm_f16<<<M_, 256, 0, stream>>>(hidden, norm_w, x16);

    // 2) in_proj: xz[M, 2048] = x16 * w16_in^T  (K=1024)
    gemm_wmma_f16<0><<<dim3(2 * DI_ / 16, M_ / 128), 256, (1024 / 32) * 1024, stream>>>(
        x16, w16_in, xz, nullptr, D_, 2 * DI_);

    // 3) causal depthwise conv + SiLU
    conv_silu<<<(M_ * DI_) / 256, 256, 0, stream>>>(xz, conv_w, conv_b, xc, xc16);

    // 4) x_proj: dbl[M, 80] = xc16 * w16_xp^T  (K=1024)
    gemm_wmma_f16<0><<<dim3((R_ + 2 * N_) / 16, M_ / 128), 256, (1024 / 32) * 1024, stream>>>(
        xc16, w16_xp, dbl, nullptr, DI_, R_ + 2 * N_);

    // 5) dt slice -> f16, then dt_proj with fused softplus(+bias): dt[M,1024] (K=64)
    prep_dt16<<<(M_ * R_) / 256, 256, 0, stream>>>(dbl, dt16);
    gemm_wmma_f16<1><<<dim3(DI_ / 16, M_ / 128), 256, (64 / 32) * 1024, stream>>>(
        dt16, w16_dt, dtbuf, dt_proj_b, R_, DI_);

    // 6) chunked selective scan: local scans -> combine -> final with gate
    scan_partial<<<(B_ * CH_ * DI_) / 256, 256, 0, stream>>>(dtbuf, xc, dbl, A_log,
                                                             chunkS, chunkSum);
    scan_combine<<<(B_ * DI_) / 256, 256, 0, stream>>>(chunkS, chunkSum, A_log, hinit);
    scan_final<<<(B_ * CH_ * DI_) / 256, 256, 0, stream>>>(dtbuf, xc, dbl, xz, A_log,
                                                           D_skip, hinit, y16);

    // 7) out_proj: out[M, 1024] = y16 * w16_out^T  (K=1024)
    gemm_wmma_f16<0><<<dim3(D_ / 16, M_ / 128), 256, (1024 / 32) * 1024, stream>>>(
        y16, w16_out, out, nullptr, DI_, D_);
}
